// DependencyParser_27315992003136
// MI455X (gfx1250) — matrix operand order repaired and checked by reading, not compiled, not run on hardware
//
#include <hip/hip_runtime.h>
#include <hip/hip_bf16.h>
#include <math.h>

#define E_ 512
#define P_ 128
#define T_ 1024

typedef __attribute__((ext_vector_type(16))) __bf16 v16bf;
typedef __attribute__((ext_vector_type(8)))  __bf16 v8bf;
typedef __attribute__((ext_vector_type(8)))  float  v8f;

__device__ __forceinline__ float sigmoidf_(float x) { return 1.0f / (1.0f + expf(-x)); }

// ---------------------------------------------------------------------------
// fp32 -> bf16 conversion (grid-stride)
// ---------------------------------------------------------------------------
__global__ void cvt_bf16_kernel(const float* __restrict__ in, __bf16* __restrict__ out, int n) {
  for (int i = blockIdx.x * blockDim.x + threadIdx.x; i < n; i += gridDim.x * blockDim.x)
    out[i] = (__bf16)in[i];
}

// ---------------------------------------------------------------------------
// Whh [2, 4H, H] fp32 -> WhhT [2, H, 4H] fp32 (so recurrence reads are
// lane-coalesced: consecutive threads read consecutive gate columns)
// ---------------------------------------------------------------------------
__global__ void transpose_whh_kernel(const float* __restrict__ in, float* __restrict__ out,
                                     int N4, int H) {
  int total = 2 * N4 * H;
  for (int idx = blockIdx.x * blockDim.x + threadIdx.x; idx < total; idx += gridDim.x * blockDim.x) {
    int d = idx / (N4 * H);
    int r = idx % (N4 * H);
    int g = r / H, k = r % H;
    out[(size_t)d * H * N4 + (size_t)k * N4 + g] = in[idx];
  }
}

// ---------------------------------------------------------------------------
// XG[dir, t, g] = sum_k X[t,k] * W[dir, g, k] + bih[dir,g] + bhh[dir,g]
// bf16 WMMA GEMM, fp32 accumulate.  C = A(row-major MxK) * B(=W^T, KxN).
// W is [N, K] row-major, so B column n lives contiguously in memory: the
// B fragment used here (lane n%16 holds one column, lane group selects the
// K half) is a single contiguous 32B load per lane.  A fragment: lane m%16
// holds row m, two contiguous 16B chunks for the two K sub-ranges.
// Block = 256 threads = 8 waves covering a 64(M) x 32(N) tile.
// ---------------------------------------------------------------------------
__global__ __launch_bounds__(256)
void gemm_xg_kernel(const __bf16* __restrict__ A, const __bf16* __restrict__ W,
                    const float* __restrict__ bih, const float* __restrict__ bhh,
                    float* __restrict__ XG, int K, int N) {
  const int dir = blockIdx.z;
  const __bf16* Wd = W + (size_t)dir * N * K;
  const float* bi = bih + (size_t)dir * N;
  const float* bh = bhh + (size_t)dir * N;
  float* out = XG + (size_t)dir * T_ * N;

  const int tid  = threadIdx.x;
  const int wave = tid >> 5, lane = tid & 31;
  const int m0 = blockIdx.x * 64 + (wave & 3) * 16;
  const int n0 = blockIdx.y * 32 + (wave >> 2) * 16;

  const int mrow = m0 + (lane & 15);
  const int kh   = (lane >> 4) * 16;   // A fragment K sub-offset
  const int ncol = n0 + (lane & 15);
  const int kg   = (lane >> 4) * 16;   // B fragment K sub-offset

  const __bf16* arow = A  + (size_t)mrow * K;
  const __bf16* wrow = Wd + (size_t)ncol * K;

  v8f acc = {};
  for (int k0 = 0; k0 < K; k0 += 32) {
    v8bf alo = *(const v8bf*)(arow + k0 + kh);
    v8bf ahi = *(const v8bf*)(arow + k0 + kh + 8);
    v16bf b  = *(const v16bf*)(wrow + k0 + kg);
    v16bf a;
#pragma unroll
    for (int i = 0; i < 8; ++i) { a[i] = alo[i]; a[8 + i] = ahi[i]; }
    acc = __builtin_amdgcn_wmma_f32_16x16x32_bf16(false, a, false, b, (short)0, acc,
                                                  false, false);
  }

  const float bsum = bi[ncol] + bh[ncol];
#pragma unroll
  for (int v = 0; v < 8; ++v) {
    const int row = m0 + v + ((lane >> 4) << 3);
    out[(size_t)row * N + ncol] = acc[v] + bsum;
  }
}

// ---------------------------------------------------------------------------
// Fused LSTM recurrences: 4 blocks = {words,pos} x {fwd,bwd}, all concurrent
// so the small pos scan hides under the words scan.  Thread u owns hidden
// unit u: computes all four gates (4 dots of length H against LDS-resident h,
// streaming WhhT from L2 with coalesced reads), then c/h update in registers.
// Only h needs sharing -> 2 barriers per timestep.
// ---------------------------------------------------------------------------
struct RecParams {
  const float* xg;    // [2, T, 4H]
  const float* wT;    // [2, H, 4H]
  float* out;         // [T, stride]
  int H;
  int stride;
  int colBase;        // column = colBase + dir*H + u
};

__global__ __launch_bounds__(512)
void lstm_rec2_kernel(RecParams pa, RecParams pb) {
  const RecParams prm = (blockIdx.x < 2) ? pa : pb;
  const int dir = blockIdx.x & 1;          // 0 = forward scan, 1 = reverse scan
  const int H = prm.H, H4 = 4 * prm.H;
  const int u = threadIdx.x;
  const bool act = (u < H);

  const float* xg = prm.xg + (size_t)dir * T_ * H4;
  const float* wT = prm.wT + (size_t)dir * H * H4;
  float* out = prm.out;

  __shared__ float h_sh[E_];               // max H = 512
  if (act) h_sh[u] = 0.0f;
  float c = 0.0f;
  __syncthreads();

  for (int t = 0; t < T_; ++t) {
    const int tt = dir ? (T_ - 1 - t) : t;
    float h = 0.0f;
    if (act) {
      float ai = 0.f, af = 0.f, ag = 0.f, ao = 0.f;
      const float* wk = wT;
#pragma unroll 4
      for (int k = 0; k < H; ++k) {
        const float hk = h_sh[k];
        ai = fmaf(wk[u],         hk, ai);
        af = fmaf(wk[H + u],     hk, af);
        ag = fmaf(wk[2 * H + u], hk, ag);
        ao = fmaf(wk[3 * H + u], hk, ao);
        wk += H4;
      }
      const float* xr = xg + (size_t)tt * H4;
      const float gi = sigmoidf_(xr[u] + ai);
      const float gf = sigmoidf_(xr[H + u] + af);
      const float gg = tanhf(xr[2 * H + u] + ag);
      const float go = sigmoidf_(xr[3 * H + u] + ao);
      c = gf * c + gi * gg;
      h = go * tanhf(c);
    }
    __syncthreads();
    if (act) h_sh[u] = h;
    __syncthreads();
    if (act) out[(size_t)tt * prm.stride + prm.colBase + dir * H + u] = h;
  }
}

// ---------------------------------------------------------------------------
// Edge scorer stage 1: s1[t] = H[t] . W1,  s2[t] = H[t] . W2  (D = 1280)
// ---------------------------------------------------------------------------
__global__ __launch_bounds__(256)
void edge_s12_kernel(const float* __restrict__ Hf, const float* __restrict__ ew,
                     float* __restrict__ s1, float* __restrict__ s2, int D) {
  const int t = blockIdx.x;
  const float* row = Hf + (size_t)t * D;
  float p1 = 0.f, p2 = 0.f;
  for (int j = threadIdx.x; j < D; j += 256) {
    const float v = row[j];
    p1 += v * ew[j];
    p2 += v * ew[D + j];
  }
  __shared__ float r1[256], r2[256];
  r1[threadIdx.x] = p1;
  r2[threadIdx.x] = p2;
  __syncthreads();
  for (int s = 128; s > 0; s >>= 1) {
    if (threadIdx.x < s) {
      r1[threadIdx.x] += r1[threadIdx.x + s];
      r2[threadIdx.x] += r2[threadIdx.x + s];
    }
    __syncthreads();
  }
  if (threadIdx.x == 0) { s1[t] = r1[0]; s2[t] = r2[0]; }
}

// ---------------------------------------------------------------------------
// Edge scorer stage 2: S[i,j] = s1[i] + s2[j] + b, diagonal zeroed
// ---------------------------------------------------------------------------
__global__ void edge_matrix_kernel(const float* __restrict__ s1, const float* __restrict__ s2,
                                   const float* __restrict__ eb, float* __restrict__ out) {
  const int idx = blockIdx.x * blockDim.x + threadIdx.x;
  if (idx >= T_ * T_) return;
  const int i = idx / T_, j = idx % T_;
  out[idx] = (i == j) ? 0.0f : (s1[i] + s2[j] + eb[0]);
}

// ---------------------------------------------------------------------------
extern "C" void kernel_launch(void* const* d_in, const int* in_sizes, int n_in,
                              void* d_out, int out_size, void* d_ws, size_t ws_size,
                              hipStream_t stream) {
  (void)in_sizes; (void)n_in; (void)out_size; (void)ws_size;

  const float* words   = (const float*)d_in[0];
  const float* pos     = (const float*)d_in[1];
  const float* w_w0_ih = (const float*)d_in[2];
  const float* w_w0_hh = (const float*)d_in[3];
  const float* w_b0_ih = (const float*)d_in[4];
  const float* w_b0_hh = (const float*)d_in[5];
  const float* w_w1_ih = (const float*)d_in[6];
  const float* w_w1_hh = (const float*)d_in[7];
  const float* w_b1_ih = (const float*)d_in[8];
  const float* w_b1_hh = (const float*)d_in[9];
  const float* p_w0_ih = (const float*)d_in[10];
  const float* p_w0_hh = (const float*)d_in[11];
  const float* p_b0_ih = (const float*)d_in[12];
  const float* p_b0_hh = (const float*)d_in[13];
  const float* p_w1_ih = (const float*)d_in[14];
  const float* p_w1_hh = (const float*)d_in[15];
  const float* p_b1_ih = (const float*)d_in[16];
  const float* p_b1_hh = (const float*)d_in[17];
  const float* edge_w  = (const float*)d_in[18];
  const float* edge_b  = (const float*)d_in[19];

  // ---- workspace carving (all offsets 256B aligned) ----
  char* ws = (char*)d_ws;
  size_t off = 0;
  auto carve = [&](size_t bytes) -> void* {
    void* p = ws + off;
    off = (off + bytes + 255) & ~(size_t)255;
    return p;
  };

  __bf16* xw_bf   = (__bf16*)carve((size_t)T_ * E_ * 2);
  __bf16* xp_bf   = (__bf16*)carve((size_t)T_ * P_ * 2);
  __bf16* xw1_bf  = (__bf16*)carve((size_t)T_ * 2 * E_ * 2);
  __bf16* xp1_bf  = (__bf16*)carve((size_t)T_ * 2 * P_ * 2);
  __bf16* w0ih_bf = (__bf16*)carve((size_t)2 * 4 * E_ * E_ * 2);
  __bf16* w1ih_bf = (__bf16*)carve((size_t)2 * 4 * E_ * 2 * E_ * 2);
  __bf16* p0ih_bf = (__bf16*)carve((size_t)2 * 4 * P_ * P_ * 2);
  __bf16* p1ih_bf = (__bf16*)carve((size_t)2 * 4 * P_ * 2 * P_ * 2);
  float* whhT_w0  = (float*)carve((size_t)2 * E_ * 4 * E_ * 4);
  float* whhT_w1  = (float*)carve((size_t)2 * E_ * 4 * E_ * 4);
  float* whhT_p0  = (float*)carve((size_t)2 * P_ * 4 * P_ * 4);
  float* whhT_p1  = (float*)carve((size_t)2 * P_ * 4 * P_ * 4);
  float* xg_w0    = (float*)carve((size_t)2 * T_ * 4 * E_ * 4);
  float* xg_w1    = (float*)carve((size_t)2 * T_ * 4 * E_ * 4);
  float* xg_p0    = (float*)carve((size_t)2 * T_ * 4 * P_ * 4);
  float* xg_p1    = (float*)carve((size_t)2 * T_ * 4 * P_ * 4);
  float* seq_w0   = (float*)carve((size_t)T_ * 2 * E_ * 4);
  float* seq_p0   = (float*)carve((size_t)T_ * 2 * P_ * 4);
  float* Hfinal   = (float*)carve((size_t)T_ * 2 * (E_ + P_) * 4);
  float* s1       = (float*)carve((size_t)T_ * 4);
  float* s2       = (float*)carve((size_t)T_ * 4);

  auto nblk = [](size_t n) { return (unsigned)((n + 255) / 256); };

  // ---- precision staging: activations/Wih -> bf16, Whh -> transposed fp32 ----
  cvt_bf16_kernel<<<nblk((size_t)T_ * E_), 256, 0, stream>>>(words, xw_bf, T_ * E_);
  cvt_bf16_kernel<<<nblk((size_t)T_ * P_), 256, 0, stream>>>(pos, xp_bf, T_ * P_);
  cvt_bf16_kernel<<<nblk((size_t)2 * 4 * E_ * E_), 256, 0, stream>>>(w_w0_ih, w0ih_bf, 2 * 4 * E_ * E_);
  cvt_bf16_kernel<<<nblk((size_t)2 * 4 * E_ * 2 * E_), 256, 0, stream>>>(w_w1_ih, w1ih_bf, 2 * 4 * E_ * 2 * E_);
  cvt_bf16_kernel<<<nblk((size_t)2 * 4 * P_ * P_), 256, 0, stream>>>(p_w0_ih, p0ih_bf, 2 * 4 * P_ * P_);
  cvt_bf16_kernel<<<nblk((size_t)2 * 4 * P_ * 2 * P_), 256, 0, stream>>>(p_w1_ih, p1ih_bf, 2 * 4 * P_ * 2 * P_);
  transpose_whh_kernel<<<nblk((size_t)2 * 4 * E_ * E_), 256, 0, stream>>>(w_w0_hh, whhT_w0, 4 * E_, E_);
  transpose_whh_kernel<<<nblk((size_t)2 * 4 * E_ * E_), 256, 0, stream>>>(w_w1_hh, whhT_w1, 4 * E_, E_);
  transpose_whh_kernel<<<nblk((size_t)2 * 4 * P_ * P_), 256, 0, stream>>>(p_w0_hh, whhT_p0, 4 * P_, P_);
  transpose_whh_kernel<<<nblk((size_t)2 * 4 * P_ * P_), 256, 0, stream>>>(p_w1_hh, whhT_p1, 4 * P_, P_);

  // ---- layer 0: input projections (WMMA) then fused recurrences ----
  gemm_xg_kernel<<<dim3(T_ / 64, (4 * E_) / 32, 2), 256, 0, stream>>>(
      xw_bf, w0ih_bf, w_b0_ih, w_b0_hh, xg_w0, E_, 4 * E_);
  gemm_xg_kernel<<<dim3(T_ / 64, (4 * P_) / 32, 2), 256, 0, stream>>>(
      xp_bf, p0ih_bf, p_b0_ih, p_b0_hh, xg_p0, P_, 4 * P_);

  RecParams rw0 = {xg_w0, whhT_w0, seq_w0, E_, 2 * E_, 0};
  RecParams rp0 = {xg_p0, whhT_p0, seq_p0, P_, 2 * P_, 0};
  lstm_rec2_kernel<<<4, 512, 0, stream>>>(rw0, rp0);

  // ---- layer 1 ----
  cvt_bf16_kernel<<<nblk((size_t)T_ * 2 * E_), 256, 0, stream>>>(seq_w0, xw1_bf, T_ * 2 * E_);
  cvt_bf16_kernel<<<nblk((size_t)T_ * 2 * P_), 256, 0, stream>>>(seq_p0, xp1_bf, T_ * 2 * P_);

  gemm_xg_kernel<<<dim3(T_ / 64, (4 * E_) / 32, 2), 256, 0, stream>>>(
      xw1_bf, w1ih_bf, w_b1_ih, w_b1_hh, xg_w1, 2 * E_, 4 * E_);
  gemm_xg_kernel<<<dim3(T_ / 64, (4 * P_) / 32, 2), 256, 0, stream>>>(
      xp1_bf, p1ih_bf, p_b1_ih, p_b1_hh, xg_p1, 2 * P_, 4 * P_);

  const int D = 2 * (E_ + P_);  // 1280
  RecParams rw1 = {xg_w1, whhT_w1, Hfinal, E_, D, 0};
  RecParams rp1 = {xg_p1, whhT_p1, Hfinal, P_, D, 2 * E_};
  lstm_rec2_kernel<<<4, 512, 0, stream>>>(rw1, rp1);

  // ---- edge scorer ----
  edge_s12_kernel<<<T_, 256, 0, stream>>>(Hfinal, edge_w, s1, s2, D);
  edge_matrix_kernel<<<(T_ * T_ + 255) / 256, 256, 0, stream>>>(s1, s2, edge_b, (float*)d_out);
}
